// MultimodalGraphSAGE_65996467470991
// MI455X (gfx1250) — matrix operand order repaired and checked by reading, not compiled, not run on hardware
//
#include <hip/hip_runtime.h>

typedef __attribute__((ext_vector_type(16))) __bf16 v16bf;
typedef __attribute__((ext_vector_type(8)))  float  v8f;

__device__ __forceinline__ unsigned short f2bf(float f) {
  unsigned u = __float_as_uint(f);
  return (unsigned short)((u + 0x7FFFu + ((u >> 16) & 1u)) >> 16);
}

// ---------------- edge aggregation (scatter-add with atomics) ----------------
__global__ void edge_agg(const float* __restrict__ x, const int* __restrict__ src,
                         const int* __restrict__ dst, float* __restrict__ agg,
                         float* __restrict__ cnt, int D) {
  int e = blockIdx.x;
  int s = src[e], d = dst[e];
  int f = threadIdx.x;
  atomicAdd(&agg[(size_t)d * D + f], x[(size_t)s * D + f]);
  if (f == 0 && cnt) atomicAdd(&cnt[d], 1.0f);
}

// ---------------- WMMA fragment packing ----------------
// A fragment (16-bit A 16x32, ISA 7.12.2): lane<16 holds K{0..7,16..23} of row M=lane,
// lane>=16 holds K{8..15,24..31} of row M=lane-16.
__device__ __forceinline__ size_t apack_idx(int m, int k, int KT) {
  int mt = m >> 4, r = m & 15, kt = k >> 5, kk = k & 31;
  int lane, i;
  if (kk < 8)       { lane = r;      i = kk;      }
  else if (kk < 16) { lane = r + 16; i = kk - 8;  }
  else if (kk < 24) { lane = r;      i = kk - 8;  }
  else              { lane = r + 16; i = kk - 16; }
  return (((size_t)mt * KT + kt) * 32 + lane) * 16 + i;
}

// Build packed A = [mean | self] in bf16 fragment order. K_total = 2*D, KT = 2*D/32.
__global__ void build_acat(const float* __restrict__ agg, const float* __restrict__ cnt,
                           const float* __restrict__ xs, unsigned short* __restrict__ Ap,
                           int Nn, int D, int KT) {
  size_t i = (size_t)blockIdx.x * blockDim.x + threadIdx.x;
  if (i >= (size_t)Nn * D) return;
  int n = (int)(i / D), f = (int)(i % D);
  float mean = agg[i] / fmaxf(cnt[n], 1.0f);
  Ap[apack_idx(n, f, KT)]     = f2bf(mean);
  Ap[apack_idx(n, D + f, KT)] = f2bf(xs[i]);
}

// B fragment (32x16): lane<16 holds K{0..7,16..23} of col N=lane; lane>=16 holds K{8..15,24..31} of col N=lane-16.
__global__ void pack_b(const float* __restrict__ B, unsigned short* __restrict__ out,
                       int K, int Nc, int ktBase, int NT) {
  size_t i = (size_t)blockIdx.x * blockDim.x + threadIdx.x;
  if (i >= (size_t)K * Nc) return;
  int k = (int)(i / Nc), n = (int)(i % Nc);
  int kt = ktBase + (k >> 5), kk = k & 31, nt = n >> 4, r = n & 15;
  int lane = r + ((kk & 8) ? 16 : 0);
  int ii = (kk & 7) + ((kk & 16) >> 1);
  out[(((size_t)kt * NT + nt) * 32 + lane) * 16 + ii] = f2bf(B[i]);
}

// ---------------- bf16 WMMA GEMM, register-blocked 16x64 per wave, double-buffered ----
// C[M x 16*NT] = Apack(ktStart..+ktCount) @ Bpack + bias. Requires NT % 4 == 0, ktCount >= 1.
__global__ void gemm_wmma_bf16(const unsigned short* __restrict__ Ap, int ktStart, int ktCount,
                               int ktStrideA, const unsigned short* __restrict__ Bp,
                               int NT, int MT, const float* __restrict__ bias,
                               float* __restrict__ C, int ldc) {
  const int ngrp = NT >> 2;                    // groups of 4 n-tiles
  int w = blockIdx.x * ((int)blockDim.x >> 5) + ((int)threadIdx.x >> 5);
  if (w >= MT * ngrp) return;                  // wave-uniform: EXEC stays all-ones
  int lane = threadIdx.x & 31;
  int mt = w / ngrp, ng = w % ngrp;
  const unsigned short* aPtr = Ap + (((size_t)mt * ktStrideA + ktStart) * 32 + lane) * 16;
  const unsigned short* bPtr = Bp + (((size_t)(ng << 2)) * 32 + lane) * 16;
  const size_t bStride = (size_t)NT * 512;     // shorts per kt step of B

  v8f acc0 = {}, acc1 = {}, acc2 = {}, acc3 = {};
  // prologue: fragments for kt = 0
  v16bf a  = *reinterpret_cast<const v16bf*>(aPtr);
  v16bf b0 = *reinterpret_cast<const v16bf*>(bPtr);
  v16bf b1 = *reinterpret_cast<const v16bf*>(bPtr + 512);
  v16bf b2 = *reinterpret_cast<const v16bf*>(bPtr + 1024);
  v16bf b3 = *reinterpret_cast<const v16bf*>(bPtr + 1536);

  for (int kt = 0; kt < ktCount - 1; ++kt) {
    aPtr += 512;
    bPtr += bStride;
    // issue next-tile loads before this tile's math so waits are partial
    v16bf an  = *reinterpret_cast<const v16bf*>(aPtr);
    v16bf bn0 = *reinterpret_cast<const v16bf*>(bPtr);
    v16bf bn1 = *reinterpret_cast<const v16bf*>(bPtr + 512);
    v16bf bn2 = *reinterpret_cast<const v16bf*>(bPtr + 1024);
    v16bf bn3 = *reinterpret_cast<const v16bf*>(bPtr + 1536);
    __builtin_prefetch(aPtr + 512, 0, 3);      // global_prefetch_b8, WGP scope: kt+2 A
    __builtin_prefetch(bPtr + bStride, 0, 3);  // global_prefetch_b8, WGP scope: kt+2 B
    acc0 = __builtin_amdgcn_wmma_f32_16x16x32_bf16(false, a, false, b0, (short)0, acc0, false, false);
    acc1 = __builtin_amdgcn_wmma_f32_16x16x32_bf16(false, a, false, b1, (short)0, acc1, false, false);
    acc2 = __builtin_amdgcn_wmma_f32_16x16x32_bf16(false, a, false, b2, (short)0, acc2, false, false);
    acc3 = __builtin_amdgcn_wmma_f32_16x16x32_bf16(false, a, false, b3, (short)0, acc3, false, false);
    a = an; b0 = bn0; b1 = bn1; b2 = bn2; b3 = bn3;
  }
  // epilogue tile
  acc0 = __builtin_amdgcn_wmma_f32_16x16x32_bf16(false, a, false, b0, (short)0, acc0, false, false);
  acc1 = __builtin_amdgcn_wmma_f32_16x16x32_bf16(false, a, false, b1, (short)0, acc1, false, false);
  acc2 = __builtin_amdgcn_wmma_f32_16x16x32_bf16(false, a, false, b2, (short)0, acc2, false, false);
  acc3 = __builtin_amdgcn_wmma_f32_16x16x32_bf16(false, a, false, b3, (short)0, acc3, false, false);

  int row0 = mt * 16 + ((lane & 16) >> 1);     // lanes 16..31 -> rows +8
  int cbase = (ng << 2) * 16 + (lane & 15);
#pragma unroll
  for (int t = 0; t < 4; ++t) {
    int col = cbase + t * 16;
    float bv = bias[col];
    const v8f& acc = (t == 0) ? acc0 : (t == 1) ? acc1 : (t == 2) ? acc2 : acc3;
#pragma unroll
    for (int r = 0; r < 8; ++r)
      C[(size_t)(row0 + r) * ldc + col] = acc[r] + bv;
  }
}

// ---------------- row L2 norm (in place) + BN column stats via atomics ----------------
__global__ void rownorm_stats(float* __restrict__ pre, float* __restrict__ colsum,
                              float* __restrict__ colsq, int C) {
  __shared__ float red[256];
  int n = blockIdx.x, c = threadIdx.x;
  float v = pre[(size_t)n * C + c];
  red[c] = v * v;
  __syncthreads();
  for (int s = C >> 1; s > 0; s >>= 1) {
    if (c < s) red[c] += red[c + s];
    __syncthreads();
  }
  float scale = 1.0f / fmaxf(sqrtf(red[0]), 1e-12f);
  v *= scale;
  pre[(size_t)n * C + c] = v;
  atomicAdd(&colsum[c], v);
  atomicAdd(&colsq[c], v * v);
}

// ---------------- BN apply + ReLU + residual add ----------------
__global__ void bn_relu_res(const float* __restrict__ pre, const float* __restrict__ res,
                            const float* __restrict__ colsum, const float* __restrict__ colsq,
                            const float* __restrict__ g, const float* __restrict__ be,
                            float* __restrict__ out, int C, int rows) {
  size_t i = (size_t)blockIdx.x * blockDim.x + threadIdx.x;
  if (i >= (size_t)rows * C) return;
  int c = (int)(i % C);
  float mu  = colsum[c] / rows;
  float var = colsq[c] / rows - mu * mu;
  float v = (pre[i] - mu) * rsqrtf(var + 1e-5f) * g[c] + be[c];
  out[i] = fmaxf(v, 0.0f) + res[i];
}

// ---------------- global mean pool ----------------
__global__ void pool_scatter(const float* __restrict__ h, const int* __restrict__ batch,
                             float* __restrict__ pooled, float* __restrict__ gcnt, int Nn) {
  size_t i = (size_t)blockIdx.x * blockDim.x + threadIdx.x;
  if (i >= (size_t)Nn * 128) return;
  int n = (int)(i / 128), c = (int)(i % 128);
  int g = batch[n];
  atomicAdd(&pooled[(size_t)g * 128 + c], h[i]);
  if (c == 0) atomicAdd(&gcnt[g], 1.0f);
}

__global__ void emb_write(const float* __restrict__ pooled, const float* __restrict__ gcnt,
                          float* __restrict__ emb, int Gq) {
  int i = blockIdx.x * blockDim.x + threadIdx.x;
  if (i >= Gq * 128) return;
  emb[i] = pooled[i] / fmaxf(gcnt[i / 128], 1.0f);
}

// ---------------- tiny classifier (scalar fp32; 64 rows, negligible cost) ----------------
__global__ void fc(const float* __restrict__ A, const float* __restrict__ W,
                   const float* __restrict__ b, float* __restrict__ Z,
                   int Gq, int K, int C) {
  int i = blockIdx.x * blockDim.x + threadIdx.x;
  if (i >= Gq * C) return;
  int g = i / C, c = i % C;
  float s = b[c];
  for (int k = 0; k < K; ++k) s += A[g * K + k] * W[k * C + c];
  Z[i] = s;
}

__global__ void bn_relu_small(float* __restrict__ Z, const float* __restrict__ g,
                              const float* __restrict__ be, int rows, int C) {
  int c = threadIdx.x;
  if (c >= C) return;
  float s = 0.f, sq = 0.f;
  for (int r = 0; r < rows; ++r) { float v = Z[r * C + c]; s += v; sq += v * v; }
  float mu = s / rows, var = sq / rows - mu * mu;
  float sc = rsqrtf(var + 1e-5f);
  for (int r = 0; r < rows; ++r) {
    float v = (Z[r * C + c] - mu) * sc * g[c] + be[c];
    Z[r * C + c] = fmaxf(v, 0.0f);
  }
}

extern "C" void kernel_launch(void* const* d_in, const int* in_sizes, int n_in,
                              void* d_out, int out_size, void* d_ws, size_t ws_size,
                              hipStream_t stream) {
  const float* x    = (const float*)d_in[0];
  const int*   ei   = (const int*)d_in[1];
  const int*   batch= (const int*)d_in[2];
  const float* Wl0  = (const float*)d_in[3];
  const float* bl0  = (const float*)d_in[4];
  const float* Wr0  = (const float*)d_in[5];
  const float* g0   = (const float*)d_in[6];
  const float* be0  = (const float*)d_in[7];
  const float* Wl1  = (const float*)d_in[8];
  const float* bl1  = (const float*)d_in[9];
  const float* Wr1  = (const float*)d_in[10];
  const float* g1   = (const float*)d_in[11];
  const float* be1  = (const float*)d_in[12];
  const float* Wp0  = (const float*)d_in[13];
  const float* bp0  = (const float*)d_in[14];
  const float* Wp1  = (const float*)d_in[15];
  const float* bp1  = (const float*)d_in[16];
  const float* Wc0  = (const float*)d_in[17];
  const float* bc0  = (const float*)d_in[18];
  const float* gc0  = (const float*)d_in[19];
  const float* bec0 = (const float*)d_in[20];
  const float* Wc1  = (const float*)d_in[21];
  const float* bc1  = (const float*)d_in[22];
  const float* gc1  = (const float*)d_in[23];
  const float* bec1 = (const float*)d_in[24];
  const float* Wc2  = (const float*)d_in[25];
  const float* bc2  = (const float*)d_in[26];

  const int Nn = in_sizes[2];       // 50000 nodes (multiple of 16)
  const int E  = in_sizes[1] / 2;   // 800000 edges
  const int Gq = 64;
  const int* src  = ei;
  const int* dstp = ei + E;
  const int MT = Nn / 16;

  // ---- carve workspace ----
  char* ws = (char*)d_ws;
  size_t cur = 0;
  auto alloc = [&](size_t bytes) -> void* {
    void* p = ws + cur;
    cur += (bytes + 255) & ~(size_t)255;
    return p;
  };
  float*          WS0    = (float*)alloc((size_t)Nn * 256 * 4);          // agg0 -> pre0 -> agg1
  unsigned short* WS1    = (unsigned short*)alloc((size_t)Nn * 512 * 2); // packed A (layer0 then layer1)
  float*          WS2    = (float*)alloc((size_t)Nn * 256 * 4);          // res0 -> pre1|res1
  float*          WS3    = (float*)alloc((size_t)Nn * 256 * 4);          // h0 -> h1
  float*          cnt    = (float*)alloc((size_t)Nn * 4);
  unsigned short* Wcat0b = (unsigned short*)alloc((size_t)8  * 16 * 512 * 2); // [Wl0;Wr0] 256x256
  unsigned short* Wp0b   = (unsigned short*)alloc((size_t)4  * 16 * 512 * 2); // 128x256
  unsigned short* Wcat1b = (unsigned short*)alloc((size_t)16 * 8  * 512 * 2); // [Wl1;Wr1] 512x128
  unsigned short* Wp1b   = (unsigned short*)alloc((size_t)8  * 8  * 512 * 2); // 256x128
  float*          stat0  = (float*)alloc(512 * 4);
  float*          stat1  = (float*)alloc(256 * 4);
  float*          pooled = (float*)alloc((size_t)Gq * 128 * 4);
  float*          gcnt   = (float*)alloc((size_t)Gq * 4);
  float*          z0     = (float*)alloc((size_t)Gq * 256 * 4);
  float*          z1     = (float*)alloc((size_t)Gq * 128 * 4);
  (void)ws_size; (void)n_in; (void)out_size;

  // ---- zero accumulators (graph-capture-legal memset nodes) ----
  hipMemsetAsync(WS0, 0, (size_t)Nn * 128 * 4, stream);   // agg0
  hipMemsetAsync(cnt, 0, (size_t)Nn * 4, stream);
  hipMemsetAsync(stat0, 0, 512 * 4, stream);
  hipMemsetAsync(stat1, 0, 256 * 4, stream);
  hipMemsetAsync(pooled, 0, (size_t)Gq * 128 * 4, stream);
  hipMemsetAsync(gcnt, 0, (size_t)Gq * 4, stream);

  // ---- pack weights to bf16 WMMA fragments ----
  pack_b<<<(128 * 256 + 255) / 256, 256, 0, stream>>>(Wl0, Wcat0b, 128, 256, 0, 16);
  pack_b<<<(128 * 256 + 255) / 256, 256, 0, stream>>>(Wr0, Wcat0b, 128, 256, 4, 16);
  pack_b<<<(128 * 256 + 255) / 256, 256, 0, stream>>>(Wp0, Wp0b,   128, 256, 0, 16);
  pack_b<<<(256 * 128 + 255) / 256, 256, 0, stream>>>(Wl1, Wcat1b, 256, 128, 0, 8);
  pack_b<<<(256 * 128 + 255) / 256, 256, 0, stream>>>(Wr1, Wcat1b, 256, 128, 8, 8);
  pack_b<<<(256 * 128 + 255) / 256, 256, 0, stream>>>(Wp1, Wp1b,   256, 128, 0, 8);

  // ---- layer 0 ----
  edge_agg<<<E, 128, 0, stream>>>(x, src, dstp, WS0, cnt, 128);
  {
    size_t tot = (size_t)Nn * 128;
    build_acat<<<(unsigned)((tot + 255) / 256), 256, 0, stream>>>(WS0, cnt, x, WS1, Nn, 128, 8);
  }
  {
    int waves = MT * (16 / 4);
    gemm_wmma_bf16<<<(waves + 3) / 4, 128, 0, stream>>>(WS1, 0, 8, 8, Wcat0b, 16, MT, bl0, WS0, 256); // pre0
    gemm_wmma_bf16<<<(waves + 3) / 4, 128, 0, stream>>>(WS1, 4, 4, 8, Wp0b,   16, MT, bp0, WS2, 256); // res0
  }
  rownorm_stats<<<Nn, 256, 0, stream>>>(WS0, stat0, stat0 + 256, 256);
  {
    size_t tot = (size_t)Nn * 256;
    bn_relu_res<<<(unsigned)((tot + 255) / 256), 256, 0, stream>>>(WS0, WS2, stat0, stat0 + 256,
                                                                   g0, be0, WS3, 256, Nn); // h0
  }

  // ---- layer 1 ----
  hipMemsetAsync(WS0, 0, (size_t)Nn * 256 * 4, stream);   // agg1
  edge_agg<<<E, 256, 0, stream>>>(WS3, src, dstp, WS0, nullptr, 256);
  {
    size_t tot = (size_t)Nn * 256;
    build_acat<<<(unsigned)((tot + 255) / 256), 256, 0, stream>>>(WS0, cnt, WS3, WS1, Nn, 256, 16);
  }
  {
    int waves = MT * (8 / 4);
    gemm_wmma_bf16<<<(waves + 3) / 4, 128, 0, stream>>>(WS1, 0, 16, 16, Wcat1b, 8, MT, bl1,
                                                        WS2, 128);                             // pre1
    gemm_wmma_bf16<<<(waves + 3) / 4, 128, 0, stream>>>(WS1, 8, 8, 16, Wp1b, 8, MT, bp1,
                                                        WS2 + (size_t)Nn * 128, 128);          // res1
  }
  rownorm_stats<<<Nn, 128, 0, stream>>>(WS2, stat1, stat1 + 128, 128);
  {
    size_t tot = (size_t)Nn * 128;
    bn_relu_res<<<(unsigned)((tot + 255) / 256), 256, 0, stream>>>(WS2, WS2 + (size_t)Nn * 128,
                                                                   stat1, stat1 + 128,
                                                                   g1, be1, WS3, 128, Nn); // h1
  }

  // ---- global mean pool + outputs ----
  {
    size_t tot = (size_t)Nn * 128;
    pool_scatter<<<(unsigned)((tot + 255) / 256), 256, 0, stream>>>(WS3, batch, pooled, gcnt, Nn);
  }
  float* out    = (float*)d_out;
  float* logits = out;        // 64 x 2
  float* emb    = out + 128;  // 64 x 128
  emb_write<<<(Gq * 128 + 255) / 256, 256, 0, stream>>>(pooled, gcnt, emb, Gq);

  // ---- classifier ----
  fc<<<(Gq * 256 + 255) / 256, 256, 0, stream>>>(emb, Wc0, bc0, z0, Gq, 128, 256);
  bn_relu_small<<<1, 256, 0, stream>>>(z0, gc0, bec0, Gq, 256);
  fc<<<(Gq * 128 + 255) / 256, 256, 0, stream>>>(z0, Wc1, bc1, z1, Gq, 256, 128);
  bn_relu_small<<<1, 128, 0, stream>>>(z1, gc1, bec1, Gq, 128);
  fc<<<(Gq * 2 + 255) / 256, 256, 0, stream>>>(z1, Wc2, bc2, logits, Gq, 128, 2);
}
